// RingAttention_22986664969035
// MI455X (gfx1250) — compile-verified, matrix operand order checked by reading, not run on hardware
//
#include <hip/hip_runtime.h>

#define BATCH  2
#define SEQLEN 2048
#define DIM    1024
#define NHEADS 16
#define DHEAD  64

typedef __attribute__((ext_vector_type(16))) _Float16 v16h;
typedef __attribute__((ext_vector_type(8)))  _Float16 v8h;
typedef __attribute__((ext_vector_type(4)))  _Float16 v4h;
typedef __attribute__((ext_vector_type(8)))  float    v8f;
typedef __attribute__((ext_vector_type(4)))  float    v4f;

// ---------------------------------------------------------------------------
// Fragment loader for the CDNA5 16x16x32 f16 WMMA A/B operand layout.
// Lane L (wave32): hi = L>>4 selects the K-slices this lane holds:
// elements [0..7] -> K = hi*8 + i, elements [8..15] -> K = 16 + hi*8 + i.
// Both slices are contiguous 16-byte chunks -> two global_load_b128.
// ---------------------------------------------------------------------------
__device__ __forceinline__ v16h load_frag(const _Float16* __restrict__ p, int hi) {
  v8h lo = *(const v8h*)(p + hi * 8);
  v8h hv = *(const v8h*)(p + 16 + hi * 8);
  return __builtin_shufflevector(lo, hv, 0,1,2,3,4,5,6,7,8,9,10,11,12,13,14,15);
}

__device__ __forceinline__ v8f wmma16(v16h a, v16h b, v8f c) {
  // D(16x16,f32) = A(16x32,f16) * B(32x16,f16) + C
  return __builtin_amdgcn_wmma_f32_16x16x32_f16(false, a, false, b, (short)0, c,
                                                false, false);
}

// ---------------------------------------------------------------------------
// f32 -> f16 conversion (vectorized by 4)
// ---------------------------------------------------------------------------
__global__ void cvt_f32_to_f16(const float* __restrict__ in,
                               _Float16* __restrict__ out, int n) {
  int i = (int)(blockIdx.x * blockDim.x + threadIdx.x) * 4;
  if (i < n) {
    v4f v = *(const v4f*)(in + i);
    v4h o;
    #pragma unroll
    for (int j = 0; j < 4; ++j) o[j] = (_Float16)v[j];
    *(v4h*)(out + i) = o;
  }
}

// ---------------------------------------------------------------------------
// QKV projection: qkv[m,n] = sum_k x[m,k] * w_qkv[n,k]   (m: B*S, n: 3*DIM)
// One wave computes a 32(m) x 64(n) tile: 8 WMMAs per 12 b128 loads.
// Q,K stored [B,S,DIM] f16; V stored transposed [B,H,DHEAD,S] f16.
// ---------------------------------------------------------------------------
__global__ void qkv_kernel(const _Float16* __restrict__ xh,
                           const _Float16* __restrict__ wh,
                           _Float16* __restrict__ Qh,
                           _Float16* __restrict__ Kh,
                           _Float16* __restrict__ Vt) {
  const int NG = (3 * DIM) / 64;                         // 48 n-groups
  int wave = (int)(blockIdx.x * blockDim.x + threadIdx.x) >> 5;
  int ng = wave % NG, mt = wave / NG;
  if (mt >= (BATCH * SEQLEN) / 32) return;
  int lane = threadIdx.x & 31, hi = lane >> 4, lm = lane & 15;
  int m0 = mt * 32, n0 = ng * 64;

  const _Float16* pa0 = xh + (size_t)(m0 + lm) * DIM;
  const _Float16* pa1 = xh + (size_t)(m0 + 16 + lm) * DIM;
  v8f acc[2][4] = {};
  for (int kc = 0; kc < DIM; kc += 32) {
    __builtin_prefetch((const void*)(pa0 + kc + 256), 0, 1);
    v16h a0 = load_frag(pa0 + kc, hi);
    v16h a1 = load_frag(pa1 + kc, hi);
    #pragma unroll
    for (int nt = 0; nt < 4; ++nt) {
      v16h b = load_frag(wh + (size_t)(n0 + nt * 16 + lm) * DIM + kc, hi);
      acc[0][nt] = wmma16(a0, b, acc[0][nt]);
      acc[1][nt] = wmma16(a1, b, acc[1][nt]);
    }
  }

  // D layout: lane holds column n; rows m = m0 + mi*16 + 8*hi + r.
  int bz = m0 / SEQLEN;
  #pragma unroll
  for (int mi = 0; mi < 2; ++mi) {
    int mrow = m0 + mi * 16 + 8 * hi;
    int sbase = (mrow - bz * SEQLEN);
    #pragma unroll
    for (int nt = 0; nt < 4; ++nt) {
      int n = n0 + nt * 16 + lm;
      if (n < DIM) {                     // Q
        #pragma unroll
        for (int r = 0; r < 8; ++r)
          Qh[(size_t)(mrow + r) * DIM + n] = (_Float16)acc[mi][nt][r];
      } else if (n < 2 * DIM) {          // K
        #pragma unroll
        for (int r = 0; r < 8; ++r)
          Kh[(size_t)(mrow + r) * DIM + (n - DIM)] = (_Float16)acc[mi][nt][r];
      } else {                           // V -> transposed, contiguous in seq
        int nn = n - 2 * DIM;
        int hh = nn >> 6, dh = nn & 63;
        v8h vv;
        #pragma unroll
        for (int r = 0; r < 8; ++r) vv[r] = (_Float16)acc[mi][nt][r];
        *(v8h*)(Vt + ((size_t)(bz * NHEADS + hh) * DHEAD + dh) * SEQLEN + sbase) = vv;
      }
    }
  }
}

// ---------------------------------------------------------------------------
// Per-query-tile online softmax over one 32-key chunk of transposed scores.
// Produces the P^T B-fragment directly from score registers and the rescale
// factor alpha for the output accumulators.
// ---------------------------------------------------------------------------
__device__ __forceinline__ void online_softmax(const v8f& s0v, const v8f& s1v,
                                               int kc, int q0t, int lm, int hi,
                                               float& m_stat, float& l,
                                               float& alpha, v16h& bp) {
  const float sm_scale = 0.125f;                         // 64^-0.5
  float s0[8], s1[8];
  #pragma unroll
  for (int r = 0; r < 8; ++r) { s0[r] = s0v[r] * sm_scale; s1[r] = s1v[r] * sm_scale; }

  if (kc + 31 > q0t) {                                   // diagonal chunk: mask
    int q = q0t + lm;
    #pragma unroll
    for (int r = 0; r < 8; ++r) {
      int key0 = kc + 8 * hi + r;
      if (key0 > q)      s0[r] = -3.0e38f;
      if (key0 + 16 > q) s1[r] = -3.0e38f;
    }
  }

  // Each query column lives in exactly 2 lanes {n, n+16}: one shfl_xor.
  float mx = -3.0e38f;
  #pragma unroll
  for (int r = 0; r < 8; ++r) mx = fmaxf(mx, fmaxf(s0[r], s1[r]));
  mx = fmaxf(mx, __shfl_xor(mx, 16, 32));
  float m_new = fmaxf(m_stat, mx);
  alpha = __expf(m_stat - m_new);
  float lsum = 0.0f;
  #pragma unroll
  for (int r = 0; r < 8; ++r) {
    s0[r] = __expf(s0[r] - m_new); lsum += s0[r];
    s1[r] = __expf(s1[r] - m_new); lsum += s1[r];
  }
  lsum += __shfl_xor(lsum, 16, 32);
  l = l * alpha + lsum;
  m_stat = m_new;
  #pragma unroll
  for (int r = 0; r < 8; ++r) { bp[r] = (_Float16)s0[r]; bp[r + 8] = (_Float16)s1[r]; }
}

// ---------------------------------------------------------------------------
// Causal flash attention, one wave per (batch, head, 32-query block).
// Scores computed transposed: S^T = K_tile * Q^T  (rows=keys, cols=queries);
// the S^T D-register layout IS the B-fragment layout for P^T in
// O^T = V^T * P^T, so probabilities feed the PV WMMA with zero data movement.
// Two 16-query tiles share every K/V fragment: 16 WMMAs per 16 b128 loads.
// ---------------------------------------------------------------------------
__global__ void attn_kernel(const _Float16* __restrict__ Qh,
                            const _Float16* __restrict__ Kh,
                            const _Float16* __restrict__ Vt,
                            _Float16* __restrict__ Oh) {
  const int QT = SEQLEN / 32;                            // 64 query blocks
  int wave = (int)(blockIdx.x * blockDim.x + threadIdx.x) >> 5;
  int qt = wave % QT;
  int h  = (wave / QT) % NHEADS;
  int bz = wave / (QT * NHEADS);
  if (bz >= BATCH) return;
  int lane = threadIdx.x & 31, hi = lane >> 4, lm = lane & 15;
  int q0 = qt * 32;

  // Loop-invariant Q^T B-fragments (lane = query column, elements = d)
  v16h bq[2][2];
  #pragma unroll
  for (int t = 0; t < 2; ++t) {
    const _Float16* qrow =
        Qh + (size_t)(bz * SEQLEN + q0 + t * 16 + lm) * DIM + h * DHEAD;
    bq[t][0] = load_frag(qrow, hi);
    bq[t][1] = load_frag(qrow + 32, hi);
  }

  const _Float16* kbase = Kh + (size_t)(bz * SEQLEN) * DIM + h * DHEAD;
  const _Float16* vbase = Vt + (size_t)(bz * NHEADS + h) * DHEAD * SEQLEN;

  v8f o[2][4] = {};
  float ms[2] = {-3.0e38f, -3.0e38f}, l[2] = {0.0f, 0.0f};

  for (int kc = 0; kc <= q0; kc += 32) {                 // causal: keys <= q0+31
    // Shared K A-fragments (two 16-key tiles x two 32-d chunks)
    const _Float16* k0p = kbase + (size_t)(kc + lm) * DIM;
    const _Float16* k1p = kbase + (size_t)(kc + 16 + lm) * DIM;
    v16h ka00 = load_frag(k0p, hi);
    v16h ka01 = load_frag(k0p + 32, hi);
    v16h ka10 = load_frag(k1p, hi);
    v16h ka11 = load_frag(k1p + 32, hi);
    // Shared V^T A-fragments (four 16-d tiles x 32 keys), contiguous loads
    v16h va[4];
    #pragma unroll
    for (int dt = 0; dt < 4; ++dt)
      va[dt] = load_frag(vbase + (size_t)(dt * 16 + lm) * SEQLEN + kc, hi);

    #pragma unroll
    for (int t = 0; t < 2; ++t) {
      v8f s0v = {}, s1v = {};
      s0v = wmma16(ka00, bq[t][0], s0v);
      s0v = wmma16(ka01, bq[t][1], s0v);
      s1v = wmma16(ka10, bq[t][0], s1v);
      s1v = wmma16(ka11, bq[t][1], s1v);

      float alpha;
      v16h bp;
      online_softmax(s0v, s1v, kc, q0 + t * 16, lm, hi, ms[t], l[t], alpha, bp);

      #pragma unroll
      for (int dt = 0; dt < 4; ++dt) {
        #pragma unroll
        for (int r = 0; r < 8; ++r) o[t][dt][r] *= alpha;
        o[t][dt] = wmma16(va[dt], bp, o[t][dt]);
      }
    }
  }

  // O^T D layout: lane holds query col lm, d = dt*16 + 8*hi + r (contiguous r)
  #pragma unroll
  for (int t = 0; t < 2; ++t) {
    float inv_l = 1.0f / l[t];
    _Float16* orow =
        Oh + (size_t)(bz * SEQLEN + q0 + t * 16 + lm) * DIM + h * DHEAD;
    #pragma unroll
    for (int dt = 0; dt < 4; ++dt) {
      v8h ov;
      #pragma unroll
      for (int r = 0; r < 8; ++r) ov[r] = (_Float16)(o[t][dt][r] * inv_l);
      *(v8h*)(orow + dt * 16 + 8 * hi) = ov;
    }
  }
}

// ---------------------------------------------------------------------------
// Output projection: out[m,n] = sum_k O[m,k] * w_o[n,k], f32 result.
// 32(m) x 64(n) per wave.
// ---------------------------------------------------------------------------
__global__ void oproj_kernel(const _Float16* __restrict__ Oh,
                             const _Float16* __restrict__ woh,
                             float* __restrict__ out) {
  const int NG = DIM / 64;                               // 16 n-groups
  int wave = (int)(blockIdx.x * blockDim.x + threadIdx.x) >> 5;
  int ng = wave % NG, mt = wave / NG;
  if (mt >= (BATCH * SEQLEN) / 32) return;
  int lane = threadIdx.x & 31, hi = lane >> 4, lm = lane & 15;
  int m0 = mt * 32, n0 = ng * 64;

  const _Float16* pa0 = Oh + (size_t)(m0 + lm) * DIM;
  const _Float16* pa1 = Oh + (size_t)(m0 + 16 + lm) * DIM;
  v8f acc[2][4] = {};
  for (int kc = 0; kc < DIM; kc += 32) {
    __builtin_prefetch((const void*)(pa0 + kc + 256), 0, 1);
    v16h a0 = load_frag(pa0 + kc, hi);
    v16h a1 = load_frag(pa1 + kc, hi);
    #pragma unroll
    for (int nt = 0; nt < 4; ++nt) {
      v16h b = load_frag(woh + (size_t)(n0 + nt * 16 + lm) * DIM + kc, hi);
      acc[0][nt] = wmma16(a0, b, acc[0][nt]);
      acc[1][nt] = wmma16(a1, b, acc[1][nt]);
    }
  }
  #pragma unroll
  for (int mi = 0; mi < 2; ++mi) {
    int mrow = m0 + mi * 16 + 8 * hi;
    #pragma unroll
    for (int nt = 0; nt < 4; ++nt) {
      int n = n0 + nt * 16 + lm;
      #pragma unroll
      for (int r = 0; r < 8; ++r)
        out[(size_t)(mrow + r) * DIM + n] = acc[mi][nt][r];
    }
  }
}

// ---------------------------------------------------------------------------
extern "C" void kernel_launch(void* const* d_in, const int* in_sizes, int n_in,
                              void* d_out, int out_size, void* d_ws, size_t ws_size,
                              hipStream_t stream) {
  (void)in_sizes; (void)n_in; (void)out_size; (void)ws_size;
  const float* x    = (const float*)d_in[0];
  const float* wqkv = (const float*)d_in[1];
  const float* wo   = (const float*)d_in[2];
  float* out = (float*)d_out;

  const size_t NX    = (size_t)BATCH * SEQLEN * DIM;     // 4,194,304
  const size_t NWQKV = (size_t)3 * DIM * DIM;            // 3,145,728
  const size_t NWO   = (size_t)DIM * DIM;                // 1,048,576

  _Float16* xh    = (_Float16*)d_ws;
  _Float16* wqkvh = xh + NX;
  _Float16* woh   = wqkvh + NWQKV;
  _Float16* Qh    = woh + NWO;
  _Float16* Kh    = Qh + NX;
  _Float16* Vt    = Kh + NX;
  _Float16* Oh    = Vt + NX;                             // total 24M halves = 48 MB

  // 1) f32 -> f16 conversions
  cvt_f32_to_f16<<<(int)(NX / 4 / 256), 256, 0, stream>>>(x, xh, (int)NX);
  cvt_f32_to_f16<<<(int)(NWQKV / 4 / 256), 256, 0, stream>>>(wqkv, wqkvh, (int)NWQKV);
  cvt_f32_to_f16<<<(int)(NWO / 4 / 256), 256, 0, stream>>>(wo, woh, (int)NWO);

  // 2) QKV projection: 128 m-tiles * 48 n-groups = 6144 waves, 8 waves/block
  qkv_kernel<<<6144 / 8, 256, 0, stream>>>(xh, wqkvh, Qh, Kh, Vt);

  // 3) Causal attention: 2*16*64 = 2048 waves
  attn_kernel<<<2048 / 8, 256, 0, stream>>>(Qh, Kh, Vt, Oh);

  // 4) Output projection: 128 m-tiles * 16 n-groups = 2048 waves
  oproj_kernel<<<2048 / 8, 256, 0, stream>>>(Oh, woh, out);
}
